// FullAttention_68899865362465
// MI455X (gfx1250) — matrix operand order repaired and checked by reading, not compiled, not run on hardware
//
#include <hip/hip_runtime.h>

// ---------------------------------------------------------------------------
// Flash attention + RoPE for MI455X (gfx1250, wave32)
// v3.1: TDM (tensor_load_to_lds, 6-arg builtin) double-buffered K/V tiles from
//       a pre-RoPE'd f16 workspace; row-sums via an extra WMMA; packed-f16
//       max butterflies.
// ---------------------------------------------------------------------------
typedef __attribute__((ext_vector_type(16))) _Float16 v16h;
typedef __attribute__((ext_vector_type(8)))  _Float16 v8h;
typedef __attribute__((ext_vector_type(2)))  _Float16 v2h;
typedef __attribute__((ext_vector_type(8)))  float    v8f;
typedef __attribute__((ext_vector_type(4)))  unsigned v4u;
typedef __attribute__((ext_vector_type(8)))  int      v8i;
typedef __attribute__((ext_vector_type(4)))  int      v4i;

#define EMBED   768
#define NHEADS  12
#define HDIM    64
#define BATCH   2
#define HH      48
#define WW      48
#define HWROWS  (HH * WW)          // 2304
#define KB      32                 // keys per block iteration
#define NKB     (HWROWS / KB)      // 72
#define QSTRIPE 128                // query rows per 256-thread block (8 waves x 16)
#define NQB     (HWROWS / QSTRIPE) // 18

#define KS_STRIDE 40               // halves: 64B data + 16B pad per row
#define VS_STRIDE 72               // halves: 128B data + 16B pad per row
#define PB_STRIDE 40

#define KT_ELEMS ((size_t)BATCH * NHEADS * HDIM * HWROWS)   // f16 workspace K

// freqs[j] = 10000^(-2j/32),  j in [0,16)
__device__ __forceinline__ float freq_of(int j) {
    return __expf((float)j * (-2.0f / 32.0f) * 9.210340371976184f);
}

// contiguous 32-byte LDS read as v16h (2x ds_load_b128)
__device__ __forceinline__ v16h ld_b256(const _Float16* p) {
    v8h lo = *(const v8h*)p;
    v8h hi = *(const v8h*)(p + 8);
    return __builtin_shufflevector(lo, hi, 0, 1, 2, 3, 4, 5, 6, 7,
                                           8, 9, 10, 11, 12, 13, 14, 15);
}

// ---- TDM: 2D tile load, global (f16 tensor) -> LDS, with LDS row padding ---
// D# per CDNA5 ISA ch.8: group0 = {count/flags, lds_addr, global_addr, type=2},
// group1 = {data_size/pad, dims, tile dims, stride}.  Groups 2/3 zero (2D).
__device__ __forceinline__ void tdm_load_2d(
    unsigned lds_off, const void* gtile,
    unsigned tensor_d0, unsigned tensor_d1,
    unsigned tile_d0,   unsigned tile_d1,
    unsigned stride0,   unsigned pad_interval, unsigned pad_amount)
{
    unsigned long long ga = (unsigned long long)(uintptr_t)gtile;
    v4u g0;
    g0[0] = 1u;                                     // count=1 (valid user D#)
    g0[1] = lds_off;                                // lds_addr (bytes)
    g0[2] = (unsigned)(ga & 0xffffffffu);           // global_addr[31:0]
    g0[3] = (unsigned)((ga >> 32) & 0x1ffffffu)     // global_addr[56:32]
          | (2u << 30);                             // type=2 ("image")
    v8i g1;
    g1[0] = (int)((1u << 16)                        // data_size=1 -> 2 bytes
                | (1u << 20)                        // pad_enable
                | (pad_interval << 22)              // dwords between pads
                | (pad_amount << 25));              // pad dwords
    g1[1] = (int)((tensor_d0 & 0xffffu) << 16);     // tensor_dim0[15:0]
    g1[2] = (int)(((tensor_d0 >> 16) & 0xffffu) | ((tensor_d1 & 0xffffu) << 16));
    g1[3] = (int)(((tensor_d1 >> 16) & 0xffffu) | (tile_d0 << 16));
    g1[4] = (int)(tile_d1 & 0xffffu);               // tile_dim1 (tile_dim2=0)
    g1[5] = (int)stride0;                           // tensor_dim0_stride[31:0]
    g1[6] = 0;
    g1[7] = 0;
    v4i z4 = {};
    v8i z8 = {};
    __builtin_amdgcn_tensor_load_to_lds(g0, g1, z4, z4, z8, 0);
}

// ---------------------------------------------------------------------------
// Prep kernels: RoPE + f32->f16 once, into workspace.
//   Kt[bh][d][pos]  (d-major: QK^T B-operand reads are contiguous)
//   Vt[bh][pos][d]  (key-major: PV B-operand reads are contiguous)
// ---------------------------------------------------------------------------
__global__ __launch_bounds__(256) void prep_k(const float* __restrict__ K,
                                              _Float16* __restrict__ Kt) {
    const int idx = blockIdx.x * 256 + threadIdx.x;  // (bh*32 + j)*HW + pos
    const int pos = idx % HWROWS;
    const int t   = idx / HWROWS;
    const int j   = t & 31;                          // rope pair
    const int bh  = t >> 5;
    const int b   = bh / NHEADS, h = bh % NHEADS;
    const float fr  = freq_of(j & 15);
    const float ang = (j < 16) ? (float)(pos / WW) * 0.5f * fr
                               : (float)(pos % WW) * 0.5f * fr;
    float sn, cs; __sincosf(ang, &sn, &cs);
    const float* kp = K + (size_t)(b * HWROWS + pos) * EMBED + h * HDIM + 2 * j;
    const float k0 = kp[0], k1 = kp[1];
    _Float16* o = Kt + ((size_t)bh * HDIM + 2 * j) * HWROWS + pos;
    o[0]      = (_Float16)(k0 * cs - k1 * sn);
    o[HWROWS] = (_Float16)(k1 * cs + k0 * sn);
}

__global__ __launch_bounds__(256) void prep_v(const float* __restrict__ V,
                                              _Float16* __restrict__ Vt) {
    const int idx  = blockIdx.x * 256 + threadIdx.x; // bh*HW*64 + pos*64 + d
    const int d    = idx & 63;
    const int rest = idx >> 6;
    const int pos  = rest % HWROWS;
    const int bh   = rest / HWROWS;
    const int b    = bh / NHEADS, h = bh % NHEADS;
    Vt[(size_t)idx] = (_Float16)V[(size_t)(b * HWROWS + pos) * EMBED + h * HDIM + d];
}

// ---------------------------------------------------------------------------
// Main flash-attention kernel
// ---------------------------------------------------------------------------
__global__ __launch_bounds__(256) void fa_rope_wmma(
    const float* __restrict__ Q, const int* __restrict__ idxs,
    const _Float16* __restrict__ Kt, const _Float16* __restrict__ Vt,
    float* __restrict__ Out)
{
    __shared__ __attribute__((aligned(16))) _Float16 ks[2][HDIM][KS_STRIDE];
    __shared__ __attribute__((aligned(16))) _Float16 vs[2][KB][VS_STRIDE];
    __shared__ __attribute__((aligned(16))) _Float16 pb[8][16][PB_STRIDE];

    const int tid    = threadIdx.x;
    const int w      = tid >> 5;
    const int lane   = tid & 31;
    const int m      = lane & 15;
    const int halfid = lane >> 4;
    const int rowb   = halfid * 8;

    int blk = blockIdx.x;
    const int qb = blk % NQB;  blk /= NQB;
    const int h  = blk % NHEADS; blk /= NHEADS;
    const int b  = blk;
    const int bh = b * NHEADS + h;

    // ---------------- Q: load + RoPE + softmax scale, f16 A-layout ----------
    const int qrow_local = qb * QSTRIPE + w * 16 + m;
    const int grow       = b * HWROWS + qrow_local;
    const int idx        = idxs[grow];
    const int qpos       = idx % HWROWS;
    const float qy = (float)(qpos / WW) * 0.5f;
    const float qx = (float)(qpos % WW) * 0.5f;
    const float* qrp = Q + (size_t)grow * EMBED + h * HDIM;

    v16h qa0, qa1;
#pragma unroll
    for (int e = 0; e < 16; ++e) {
        const int kl = (e < 8) ? (halfid * 8 + e) : (halfid * 8 + e + 8);
#pragma unroll
        for (int s = 0; s < 2; ++s) {
            const int d = s * 32 + kl;
            const int j = d >> 1;
            const float fr  = freq_of(j & 15);
            const float ang = (j < 16) ? qy * fr : qx * fr;
            float sn, cs; __sincosf(ang, &sn, &cs);
            const float q0 = qrp[2 * j], q1 = qrp[2 * j + 1];
            const float val = (d & 1) ? (q1 * cs + q0 * sn)
                                      : (q0 * cs - q1 * sn);
            if (s == 0) qa0[e] = (_Float16)(val * 0.125f);
            else        qa1[e] = (_Float16)(val * 0.125f);
        }
    }

    v16h ones;
#pragma unroll
    for (int e = 0; e < 16; ++e) ones[e] = (_Float16)1.0f;

    v8f o0 = {}, o1 = {}, o2 = {}, o3 = {};
    float row_max[8], row_sum[8];
#pragma unroll
    for (int r = 0; r < 8; ++r) { row_max[r] = -3.0e38f; row_sum[r] = 0.0f; }

    const _Float16* ktb = Kt + (size_t)bh * HDIM * HWROWS;  // [64][2304]
    const _Float16* vtb = Vt + (size_t)bh * HWROWS * HDIM;  // [2304][64]
    const unsigned ks_off0 = (unsigned)(uintptr_t)&ks[0][0][0];
    const unsigned ks_off1 = (unsigned)(uintptr_t)&ks[1][0][0];
    const unsigned vs_off0 = (unsigned)(uintptr_t)&vs[0][0][0];
    const unsigned vs_off1 = (unsigned)(uintptr_t)&vs[1][0][0];

    // prologue: TDM-load tile 0 into buffer 0 (wave 0 only; TDM ignores EXEC)
    if (w == 0) {
        tdm_load_2d(ks_off0, ktb, HWROWS, HDIM, KB, HDIM, HWROWS, 3u, 3u);
        tdm_load_2d(vs_off0, vtb, HDIM, HWROWS, HDIM, KB, HDIM, 4u, 3u);
    }

    for (int kblk = 0; kblk < NKB; ++kblk) {
        const int buf = kblk & 1;
        // issue next tile into the other buffer, then wait for current tile.
        // TDM ops are in-order per wave: tensorcnt<=2 covers tile kblk.
        if (w == 0) {
            if (kblk + 1 < NKB) {
                const int nb = buf ^ 1;
                tdm_load_2d(nb ? ks_off1 : ks_off0,
                            ktb + (size_t)(kblk + 1) * KB,
                            HWROWS, HDIM, KB, HDIM, HWROWS, 3u, 3u);
                tdm_load_2d(nb ? vs_off1 : vs_off0,
                            vtb + (size_t)(kblk + 1) * KB * HDIM,
                            HDIM, HWROWS, HDIM, KB, HDIM, 4u, 3u);
                __builtin_amdgcn_s_wait_tensorcnt(2);
            } else {
                __builtin_amdgcn_s_wait_tensorcnt(0);
            }
        }
        __syncthreads();   // tile kblk visible to all waves

        // ---- S = Q x K^T : two 16x16 f32 tiles -----------------------------
        v8f s0, s1;
        {
            v16h b0, b1, b2, b3;
#pragma unroll
            for (int e = 0; e < 16; ++e) {
                const int kl = (e < 8) ? (halfid * 8 + e) : (halfid * 8 + e + 8);
                b0[e] = ks[buf][kl][m];
                b1[e] = ks[buf][32 + kl][m];
                b2[e] = ks[buf][kl][16 + m];
                b3[e] = ks[buf][32 + kl][16 + m];
            }
            v8f cz = {};
            s0 = __builtin_amdgcn_wmma_f32_16x16x32_f16(false, qa0, false, b0, (short)0, cz, false, false);
            s0 = __builtin_amdgcn_wmma_f32_16x16x32_f16(false, qa1, false, b1, (short)0, s0, false, false);
            v8f cz1 = {};
            s1 = __builtin_amdgcn_wmma_f32_16x16x32_f16(false, qa0, false, b2, (short)0, cz1, false, false);
            s1 = __builtin_amdgcn_wmma_f32_16x16x32_f16(false, qa1, false, b3, (short)0, s1, false, false);
        }

        // ---- block row-max: packed f16x2 butterfly over 16 lanes -----------
        v2h pk[4];
#pragma unroll
        for (int i = 0; i < 4; ++i) {
            pk[i][0] = (_Float16)fmaxf(s0[2 * i],     s1[2 * i]);
            pk[i][1] = (_Float16)fmaxf(s0[2 * i + 1], s1[2 * i + 1]);
        }
#pragma unroll
        for (int msk = 1; msk < 16; msk <<= 1)
#pragma unroll
            for (int i = 0; i < 4; ++i) {
                const int x = __shfl_xor(__builtin_bit_cast(int, pk[i]), msk, 32);
                pk[i] = __builtin_elementwise_max(pk[i], __builtin_bit_cast(v2h, x));
            }

        float alpha[8], p0[8], p1[8];
#pragma unroll
        for (int r = 0; r < 8; ++r) {
            const float bm = (float)pk[r >> 1][r & 1];
            const float mn = fmaxf(row_max[r], bm);
            alpha[r]   = __expf(row_max[r] - mn);
            row_max[r] = mn;
            p0[r] = __expf(s0[r] - mn);
            p1[r] = __expf(s1[r] - mn);
            o0[r] *= alpha[r]; o1[r] *= alpha[r];
            o2[r] *= alpha[r]; o3[r] *= alpha[r];
        }

        // ---- P: C-layout -> per-wave LDS -> A-layout f16 -------------------
#pragma unroll
        for (int r = 0; r < 8; ++r) {
            pb[w][rowb + r][m]      = (_Float16)p0[r];
            pb[w][rowb + r][16 + m] = (_Float16)p1[r];
        }
        __syncthreads();

        v16h pa;
        {
            v8h lo = *(const v8h*)&pb[w][m][halfid * 8];
            v8h hi = *(const v8h*)&pb[w][m][16 + halfid * 8];
            pa = __builtin_shufflevector(lo, hi, 0, 1, 2, 3, 4, 5, 6, 7,
                                                 8, 9, 10, 11, 12, 13, 14, 15);
        }

        // ---- row sums via WMMA against all-ones B (replaces butterfly) -----
        v8f czs = {};
        v8f psum = __builtin_amdgcn_wmma_f32_16x16x32_f16(false, pa, false,
                       ones, (short)0, czs, false, false);
#pragma unroll
        for (int r = 0; r < 8; ++r)
            row_sum[r] = row_sum[r] * alpha[r] + psum[r];

        // ---- O += P x V ----------------------------------------------------
        v16h vb0, vb1, vb2, vb3;
#pragma unroll
        for (int e = 0; e < 16; ++e) {
            const int kl = (e < 8) ? (halfid * 8 + e) : (halfid * 8 + e + 8);
            vb0[e] = vs[buf][kl][m];
            vb1[e] = vs[buf][kl][16 + m];
            vb2[e] = vs[buf][kl][32 + m];
            vb3[e] = vs[buf][kl][48 + m];
        }
        o0 = __builtin_amdgcn_wmma_f32_16x16x32_f16(false, pa, false,
                 vb0, (short)0, o0, false, false);
        o1 = __builtin_amdgcn_wmma_f32_16x16x32_f16(false, pa, false,
                 vb1, (short)0, o1, false, false);
        o2 = __builtin_amdgcn_wmma_f32_16x16x32_f16(false, pa, false,
                 vb2, (short)0, o2, false, false);
        o3 = __builtin_amdgcn_wmma_f32_16x16x32_f16(false, pa, false,
                 vb3, (short)0, o3, false, false);

        __syncthreads();   // all reads of buf done before it is refilled
    }

    // ---------------- finalize ----------------------------------------------
    const size_t obase = (size_t)(b * HWROWS + qb * QSTRIPE + w * 16) * EMBED + h * HDIM;
#pragma unroll
    for (int r = 0; r < 8; ++r) {
        const float inv = 1.0f / row_sum[r];
        float* orow = Out + obase + (size_t)(rowb + r) * EMBED + m;
        orow[0]  = o0[r] * inv;
        orow[16] = o1[r] * inv;
        orow[32] = o2[r] * inv;
        orow[48] = o3[r] * inv;
    }
}

extern "C" void kernel_launch(void* const* d_in, const int* in_sizes, int n_in,
                              void* d_out, int out_size, void* d_ws, size_t ws_size,
                              hipStream_t stream) {
    const float* q    = (const float*)d_in[0];
    const float* k    = (const float*)d_in[1];
    const float* v    = (const float*)d_in[2];
    const int*   idxs = (const int*)d_in[3];
    float* out = (float*)d_out;

    _Float16* Kt = (_Float16*)d_ws;            // [24][64][2304] f16
    _Float16* Vt = Kt + KT_ELEMS;              // [24][2304][64] f16

    hipLaunchKernelGGL(prep_k, dim3(BATCH * NHEADS * 32 * HWROWS / 256), dim3(256),
                       0, stream, k, Kt);
    hipLaunchKernelGGL(prep_v, dim3(BATCH * NHEADS * HWROWS * HDIM / 256), dim3(256),
                       0, stream, v, Vt);
    hipLaunchKernelGGL(fa_rope_wmma, dim3(BATCH * NHEADS * NQB), dim3(256),
                       0, stream, q, idxs, Kt, Vt, out);
}
